// MultiheadedSelfAttention_30940944400400
// MI455X (gfx1250) — matrix operand order repaired and checked by reading, not compiled, not run on hardware
//
#include <hip/hip_runtime.h>
#include <hip/hip_bf16.h>

// ---------------------------------------------------------------------------
// Multi-head self-attention (GPT-2 small shape): B=4, T=2048, D=768, H=12.
// f16 WMMA (V_WMMA_F32_16X16X32_F16) everywhere, fp32 accumulate.
// Inputs pre-converted to f16 once; GEMM loops double-buffered + unrolled x2
// so prefetch loads overlap WMMAs with no register-rotation copies.
// ---------------------------------------------------------------------------

typedef _Float16 h16 __attribute__((ext_vector_type(16)));
typedef _Float16 h8  __attribute__((ext_vector_type(8)));
typedef float    f8  __attribute__((ext_vector_type(8)));

#define DMODEL 768
#define NHEAD  12
#define DH     64
#define TSEQ   2048
#define NBATCH 4

__device__ __forceinline__ f8 wmma_f16(h16 a, h16 b, f8 c) {
  // (neg_a, A, neg_b, B, c_mod, C, reuse_a, reuse_b)
  return __builtin_amdgcn_wmma_f32_16x16x32_f16(false, a, false, b, (short)0, c,
                                                false, false);
}

// ---- Fragment loaders -------------------------------------------------------
// A matrix 16x32 f16, row-major source, row stride ld (elements).
// Lane L (L&15 = row), kh=(lane>>4): e=0..7 -> K=kh*8+e ; e=8..15 -> K=16+kh*8+(e-8).
__device__ __forceinline__ h16 load_a_f16(const _Float16* __restrict__ base, int ld) {
  const int lane = threadIdx.x & 31;
  const _Float16* p = base + (size_t)(lane & 15) * ld + ((lane >> 4) << 3);
  h8 lo = *(const h8*)(p);
  h8 hi = *(const h8*)(p + 16);
  h16 r;
#pragma unroll
  for (int i = 0; i < 8; ++i) { r[i] = lo[i]; r[i + 8] = hi[i]; }
  return r;
}

// B matrix 32x16 (KxN) with B[k][n] = src[n][k]; src row-major, row stride ld.
// Lane layout: col = lane&15, kh=(lane>>4): e=0..7 -> K=kh*8+e ; e=8..15 -> K=16+kh*8+(e-8).
__device__ __forceinline__ h16 load_b_f16(const _Float16* __restrict__ base, int ld) {
  const int lane = threadIdx.x & 31;
  const _Float16* p = base + (size_t)(lane & 15) * ld + ((lane >> 4) << 3);
  h8 lo = *(const h8*)(p);
  h8 hi = *(const h8*)(p + 16);
  h16 r;
#pragma unroll
  for (int i = 0; i < 8; ++i) { r[i] = lo[i]; r[i + 8] = hi[i]; }
  return r;
}

// ---------------------------------------------------------------------------
// Kernel 0: one-shot f32 -> f16 conversion (vectorized, 8 elems/thread).
// ---------------------------------------------------------------------------
__global__ __launch_bounds__(256) void cvt_f32_to_f16(
    const float* __restrict__ src, _Float16* __restrict__ dst, int n8) {
  const int i = blockIdx.x * 256 + threadIdx.x;
  if (i < n8) {
    f8 v = ((const f8*)src)[i];
    h8 o;
#pragma unroll
    for (int j = 0; j < 8; ++j) o[j] = (_Float16)v[j];
    ((h8*)dst)[i] = o;
  }
}

// ---------------------------------------------------------------------------
// Double-buffered 16x64-per-wave GEMM body: acc[s] += A(16x768) x B(768x64).
// A row-major [*, 768] f16; B from W rows n0..n0+63 (B[k][n] = W[n][k]).
// Unrolled x2: buffers alternate roles -> no register rotation copies.
// ---------------------------------------------------------------------------
__device__ __forceinline__ void gemm_strip(const _Float16* __restrict__ Arow,
                                           const _Float16* __restrict__ Wrow,
                                           f8 acc[4]) {
  h16 a0, a1, b0[4], b1[4];
  a0 = load_a_f16(Arow, DMODEL);
  a1 = load_a_f16(Arow + 32, DMODEL);
#pragma unroll
  for (int s = 0; s < 4; ++s) {
    b0[s] = load_b_f16(Wrow + (size_t)s * 16 * DMODEL, DMODEL);
    b1[s] = load_b_f16(Wrow + (size_t)s * 16 * DMODEL + 32, DMODEL);
  }

  // 24 K-steps of 32: buf0 handles even steps, buf1 odd steps.
  for (int k0 = 0; k0 < DMODEL - 96; k0 += 64) {
    // consume buf0 (step k0), then refill it for step k0+64
#pragma unroll
    for (int s = 0; s < 4; ++s) acc[s] = wmma_f16(a0, b0[s], acc[s]);
    a0 = load_a_f16(Arow + k0 + 64, DMODEL);
#pragma unroll
    for (int s = 0; s < 4; ++s)
      b0[s] = load_b_f16(Wrow + (size_t)s * 16 * DMODEL + k0 + 64, DMODEL);
    // consume buf1 (step k0+32), then refill it for step k0+96
#pragma unroll
    for (int s = 0; s < 4; ++s) acc[s] = wmma_f16(a1, b1[s], acc[s]);
    a1 = load_a_f16(Arow + k0 + 96, DMODEL);
#pragma unroll
    for (int s = 0; s < 4; ++s)
      b1[s] = load_b_f16(Wrow + (size_t)s * 16 * DMODEL + k0 + 96, DMODEL);
  }
  // epilogue: steps 704 (buf0) and 736 (buf1)
#pragma unroll
  for (int s = 0; s < 4; ++s) acc[s] = wmma_f16(a0, b0[s], acc[s]);
#pragma unroll
  for (int s = 0; s < 4; ++s) acc[s] = wmma_f16(a1, b1[s], acc[s]);
}

// ---------------------------------------------------------------------------
// Kernel 1: QKV projection.  out = x @ W.T + b  (W row n = output feature n)
// grid: (B*T/64, 12 head tiles, 3 matrices), block: 128 (4 waves).
// Q,K stored [BH][T][Dh] f16 (Q pre-scaled by 1/8); V stored transposed
// [BH][Dh][T] f16 so P·V B-fragments are contiguous.
// ---------------------------------------------------------------------------
__global__ __launch_bounds__(128) void qkv_kernel(
    const _Float16* __restrict__ Xh,
    const _Float16* __restrict__ Wqh, const float* __restrict__ bq,
    const _Float16* __restrict__ Wkh, const float* __restrict__ bk,
    const _Float16* __restrict__ Wvh, const float* __restrict__ bv,
    _Float16* __restrict__ Qo, _Float16* __restrict__ Ko,
    _Float16* __restrict__ Vt) {
  const int wave = threadIdx.x >> 5;
  const int lane = threadIdx.x & 31;
  const int m0 = blockIdx.x * 64 + wave * 16;  // global row into [B*T]
  const int h = blockIdx.y;
  const int mat = blockIdx.z;

  const _Float16* W = (mat == 0) ? Wqh : (mat == 1) ? Wkh : Wvh;
  const float* bias = (mat == 0) ? bq : (mat == 1) ? bk : bv;
  const float scale = (mat == 0) ? 0.125f : 1.0f;  // 1/sqrt(64) folded into Q

  f8 acc[4] = {};
  gemm_strip(Xh + (size_t)m0 * DMODEL, W + (size_t)h * DH * DMODEL, acc);

  const int col = lane & 15;
  const int ro = (lane >> 4) * 8;
#pragma unroll
  for (int s = 0; s < 4; ++s) {
    const int dh = s * 16 + col;
    const float bb = bias[h * DH + dh];
#pragma unroll
    for (int r = 0; r < 8; ++r) {
      const int gid = m0 + r + ro;          // row in [B*T]
      const int bidx = gid >> 11;           // /2048
      const int t = gid & (TSEQ - 1);
      const float v = (acc[s][r] + bb) * scale;
      const size_t bh = (size_t)bidx * NHEAD + h;
      if (mat == 2) {
        Vt[(bh * DH + dh) * TSEQ + t] = (_Float16)v;
      } else if (mat == 0) {
        Qo[(bh * TSEQ + t) * DH + dh] = (_Float16)v;
      } else {
        Ko[(bh * TSEQ + t) * DH + dh] = (_Float16)v;
      }
    }
  }
}

// ---------------------------------------------------------------------------
// Kernel 2: causal flash attention.  grid: (T/64, B*H), block 128 (4 waves).
// Each wave owns 16 query rows; streams key tiles of 64 with online softmax.
// V fragments prefetched before the softmax VALU work to overlap latency.
// ---------------------------------------------------------------------------
__global__ __launch_bounds__(128) void attn_kernel(
    const _Float16* __restrict__ Q, const _Float16* __restrict__ Km,
    const _Float16* __restrict__ Vt, _Float16* __restrict__ Y) {
  __shared__ _Float16 Plds[4][16][72];  // per-wave P staging (pad to 72)

  const int wave = threadIdx.x >> 5;
  const int lane = threadIdx.x & 31;
  const int q0 = blockIdx.x * 64;
  const int bh = blockIdx.y;
  const int bidx = bh / NHEAD;
  const int h = bh % NHEAD;
  const int col = lane & 15;
  const int ro = (lane >> 4) * 8;

  // Preload the two Q A-fragments (K-dim = Dh = 64 -> two 32-slices).
  const _Float16* Qb = Q + ((size_t)bh * TSEQ + q0 + wave * 16) * DH;
  h16 qa0 = load_a_f16(Qb, DH);
  h16 qa1 = load_a_f16(Qb + 32, DH);

  f8 o[4] = {};
  float mrun[8], lrun[8];
#pragma unroll
  for (int r = 0; r < 8; ++r) { mrun[r] = -1e30f; lrun[r] = 0.0f; }

  const int nkt = blockIdx.x + 1;  // causal: key tiles 0..qtile
  for (int kt = 0; kt < nkt; ++kt) {
    // ---- S = Q . K^T  (16x64 strip): batch 8 K-frag loads, then 8 WMMAs ----
    const _Float16* Kb = Km + ((size_t)bh * TSEQ + kt * 64) * DH;
    h16 kb[8];
#pragma unroll
    for (int sub = 0; sub < 4; ++sub) {
      kb[2 * sub]     = load_b_f16(Kb + (size_t)sub * 16 * DH, DH);
      kb[2 * sub + 1] = load_b_f16(Kb + (size_t)sub * 16 * DH + 32, DH);
    }
    f8 s[4] = {};
#pragma unroll
    for (int sub = 0; sub < 4; ++sub) {
      s[sub] = wmma_f16(qa0, kb[2 * sub], s[sub]);
      s[sub] = wmma_f16(qa1, kb[2 * sub + 1], s[sub]);
    }

    // ---- prefetch V fragments; they overlap the softmax VALU below ----
    const _Float16* Vb = Vt + (size_t)bh * DH * TSEQ + kt * 64;
    h16 vb[8];
#pragma unroll
    for (int ks = 0; ks < 2; ++ks)
#pragma unroll
      for (int sub = 0; sub < 4; ++sub)
        vb[ks * 4 + sub] =
            load_b_f16(Vb + (size_t)sub * 16 * TSEQ + ks * 32, TSEQ);

    // ---- causal mask on the diagonal tile ----
    if (kt == (int)blockIdx.x) {
#pragma unroll
      for (int sub = 0; sub < 4; ++sub) {
        const int j = kt * 64 + sub * 16 + col;
#pragma unroll
        for (int r = 0; r < 8; ++r) {
          const int trow = q0 + wave * 16 + r + ro;
          if (j > trow) s[sub][r] = -1e30f;
        }
      }
    }

    // ---- online softmax (row stats via 16-lane xor reductions) ----
    float mnew[8], alpha[8];
#pragma unroll
    for (int r = 0; r < 8; ++r) {
      float v = fmaxf(fmaxf(s[0][r], s[1][r]), fmaxf(s[2][r], s[3][r]));
      v = fmaxf(v, __shfl_xor(v, 1, 32));
      v = fmaxf(v, __shfl_xor(v, 2, 32));
      v = fmaxf(v, __shfl_xor(v, 4, 32));
      v = fmaxf(v, __shfl_xor(v, 8, 32));
      mnew[r] = fmaxf(mrun[r], v);
      alpha[r] = __expf(mrun[r] - mnew[r]);
      mrun[r] = mnew[r];
    }
#pragma unroll
    for (int sub = 0; sub < 4; ++sub)
#pragma unroll
      for (int r = 0; r < 8; ++r) s[sub][r] = __expf(s[sub][r] - mnew[r]);
#pragma unroll
    for (int r = 0; r < 8; ++r) {
      float rs = s[0][r] + s[1][r] + s[2][r] + s[3][r];
      rs += __shfl_xor(rs, 1, 32);
      rs += __shfl_xor(rs, 2, 32);
      rs += __shfl_xor(rs, 4, 32);
      rs += __shfl_xor(rs, 8, 32);
      lrun[r] = lrun[r] * alpha[r] + rs;
    }
#pragma unroll
    for (int sub = 0; sub < 4; ++sub)
#pragma unroll
      for (int r = 0; r < 8; ++r) o[sub][r] *= alpha[r];

    // ---- stage P (f16) through per-wave LDS to re-layout as A-fragment ----
#pragma unroll
    for (int sub = 0; sub < 4; ++sub)
#pragma unroll
      for (int r = 0; r < 8; ++r)
        Plds[wave][r + ro][sub * 16 + col] = (_Float16)s[sub][r];
    asm volatile("s_wait_dscnt 0" ::: "memory");  // wave-local LDS RAW fence

    // ---- O += P . V ----
#pragma unroll
    for (int ks = 0; ks < 2; ++ks) {
      h16 pa = load_a_f16(&Plds[wave][0][ks * 32], 72);
#pragma unroll
      for (int sub = 0; sub < 4; ++sub)
        o[sub] = wmma_f16(pa, vb[ks * 4 + sub], o[sub]);
    }
    asm volatile("s_wait_dscnt 0" ::: "memory");  // WAR fence before next tile
  }

  // ---- normalize and write Y [B][T][H*Dh] f16 ----
#pragma unroll
  for (int sub = 0; sub < 4; ++sub) {
    const int dh = sub * 16 + col;
#pragma unroll
    for (int r = 0; r < 8; ++r) {
      const int t = q0 + wave * 16 + r + ro;
      const float v = o[sub][r] / lrun[r];
      Y[((size_t)bidx * TSEQ + t) * DMODEL + h * DH + dh] = (_Float16)v;
    }
  }
}

// ---------------------------------------------------------------------------
// Kernel 3: output projection.  out = Y @ Wp.T + bp   (fp32 output)
// grid: (B*T/64, 12), block 128 (4 waves).
// ---------------------------------------------------------------------------
__global__ __launch_bounds__(128) void proj_kernel(
    const _Float16* __restrict__ Y, const _Float16* __restrict__ Wph,
    const float* __restrict__ bp, float* __restrict__ out) {
  const int wave = threadIdx.x >> 5;
  const int lane = threadIdx.x & 31;
  const int m0 = blockIdx.x * 64 + wave * 16;
  const int n0 = blockIdx.y * 64;

  f8 acc[4] = {};
  gemm_strip(Y + (size_t)m0 * DMODEL, Wph + (size_t)n0 * DMODEL, acc);

  const int col = lane & 15;
  const int ro = (lane >> 4) * 8;
#pragma unroll
  for (int s = 0; s < 4; ++s) {
    const int n = n0 + s * 16 + col;
    const float bb = bp[n];
#pragma unroll
    for (int r = 0; r < 8; ++r)
      out[(size_t)(m0 + r + ro) * DMODEL + n] = acc[s][r] + bb;
  }
}

// ---------------------------------------------------------------------------
extern "C" void kernel_launch(void* const* d_in, const int* in_sizes, int n_in,
                              void* d_out, int out_size, void* d_ws,
                              size_t ws_size, hipStream_t stream) {
  const float* x  = (const float*)d_in[0];
  const float* Wq = (const float*)d_in[1];
  const float* bq = (const float*)d_in[2];
  const float* Wk = (const float*)d_in[3];
  const float* bk = (const float*)d_in[4];
  const float* Wv = (const float*)d_in[5];
  const float* bv = (const float*)d_in[6];
  const float* Wp = (const float*)d_in[7];
  const float* bp = (const float*)d_in[8];
  float* out = (float*)d_out;

  const size_t SZ = (size_t)NBATCH * NHEAD * TSEQ * DH;  // 6,291,456 elems
  const size_t WSZ = (size_t)DMODEL * DMODEL;            //   589,824 elems
  _Float16* Q   = (_Float16*)d_ws;
  _Float16* K   = Q + SZ;
  _Float16* Vt  = K + SZ;
  _Float16* Y   = Vt + SZ;
  _Float16* Xh  = Y + SZ;
  _Float16* Wqh = Xh + SZ;
  _Float16* Wkh = Wqh + WSZ;
  _Float16* Wvh = Wkh + WSZ;
  _Float16* Wph = Wvh + WSZ;

  dim3 blk(128, 1, 1);
  dim3 cblk(256, 1, 1);

  // One-shot f32->f16 conversions (memory-bound, done once).
  {
    const int x8 = (int)(SZ / 8);   // 786,432
    const int w8 = (int)(WSZ / 8);  //  73,728
    cvt_f32_to_f16<<<dim3((x8 + 255) / 256), cblk, 0, stream>>>(x, Xh, x8);
    cvt_f32_to_f16<<<dim3((w8 + 255) / 256), cblk, 0, stream>>>(Wq, Wqh, w8);
    cvt_f32_to_f16<<<dim3((w8 + 255) / 256), cblk, 0, stream>>>(Wk, Wkh, w8);
    cvt_f32_to_f16<<<dim3((w8 + 255) / 256), cblk, 0, stream>>>(Wv, Wvh, w8);
    cvt_f32_to_f16<<<dim3((w8 + 255) / 256), cblk, 0, stream>>>(Wp, Wph, w8);
  }

  dim3 g1((NBATCH * TSEQ) / 64, DMODEL / 64, 3);
  qkv_kernel<<<g1, blk, 0, stream>>>(Xh, Wqh, bq, Wkh, bk, Wvh, bv, Q, K, Vt);

  dim3 g2(TSEQ / 64, NBATCH * NHEAD, 1);
  attn_kernel<<<g2, blk, 0, stream>>>(Q, K, Vt, Y);

  dim3 g3((NBATCH * TSEQ) / 64, DMODEL / 64, 1);
  proj_kernel<<<g3, blk, 0, stream>>>(Y, Wph, bp, out);
}
